// MPNNEncoder_33303176413480
// MI455X (gfx1250) — matrix operand order repaired and checked by reading, not compiled, not run on hardware
//
#include <hip/hip_runtime.h>
#include <hip/hip_bf16.h>

// ---------------------------------------------------------------------------
// CMPNN / MPNNEncoder forward for gfx1250 (MI455X, wave32, WMMA).
// GEMMs: branch-free WMMA f32_16x16x32_f16 tiles over zero-padded f16 operands.
// ---------------------------------------------------------------------------

typedef __attribute__((ext_vector_type(16))) _Float16 v16h;
typedef __attribute__((ext_vector_type(8)))  _Float16 v8h;
typedef __attribute__((ext_vector_type(8)))  float    v8f;

#define H_DIM   300
#define H3_DIM  900
#define N_MOLS  2048
#define L_SEQ   32
#define N_ATOMS 65537
#define N_BONDS 262145
#define MAX_NB  6

// padded sizes (M -> x64, K -> x32, weight-N -> x64)
#define A_PAD   65600      // ceil(65537/64)*64
#define E_PAD   262208     // ceil(262145/64)*64
#define BL_SEQ  65536      // 2048*32, already x64
#define K133P   160
#define K147P   160
#define K300P   320
#define K600P   608
#define K900P   928
#define N300P   320
#define N900P   960

union hvec16 { v16h v; v8h h[2]; };

// ---------------------------------------------------------------------------
// Branch-free WMMA GEMM: C[M,N] = act(A16[Mp,lda] @ W16[Np,ldw]^T + bias + addend)
// A16/W16 zero-padded f16; one wave = 16x64 tile; K-loop step 32.
// ---------------------------------------------------------------------------
__global__ __launch_bounds__(128)
void wmma_gemm2_kernel(const _Float16* __restrict__ A16, int lda,
                       const _Float16* __restrict__ W16, int ldw,
                       const float* __restrict__ bias,    // [N] or nullptr
                       const float* __restrict__ addend,  // [M,N] or nullptr
                       float* __restrict__ C,             // [M,N] fp32
                       int M, int N, int Kp, int relu_flag) {
  const int wave = threadIdx.x >> 5;
  const int lane = threadIdx.x & 31;
  const int m0 = (blockIdx.y * 4 + wave) * 16;
  const int n0 = blockIdx.x * 64;
  if (m0 >= M) return;                 // wave-uniform: EXEC stays all-ones

  const int half_id = lane >> 4;       // 0: lanes 0-15, 1: lanes 16-31
  const int lr      = lane & 15;

  // A fragment: row m0+lr; lanes 0-15 take K 0-15, lanes 16-31 K 16-31
  const _Float16* ap = A16 + (size_t)(m0 + lr) * lda + half_id * 16;
  // B fragment: column n0+16t+lr of W; lanes 0-15 K 0-15, lanes 16-31 K 16-31
  const _Float16* bp = W16 + (size_t)(n0 + lr) * ldw + half_id * 16;
  const size_t bstep = (size_t)16 * ldw;

  v8f acc0 = {}, acc1 = {}, acc2 = {}, acc3 = {};
  for (int k0 = 0; k0 < Kp; k0 += 32) {
    hvec16 af;
    af.h[0] = *(const v8h*)(ap + k0);
    af.h[1] = *(const v8h*)(ap + k0 + 8);
    const v16h b0 = *(const v16h*)(bp + k0);
    const v16h b1 = *(const v16h*)(bp + bstep + k0);
    const v16h b2 = *(const v16h*)(bp + 2 * bstep + k0);
    const v16h b3 = *(const v16h*)(bp + 3 * bstep + k0);
    acc0 = __builtin_amdgcn_wmma_f32_16x16x32_f16(false, af.v, false, b0,
                                                  (short)0, acc0, false, false);
    acc1 = __builtin_amdgcn_wmma_f32_16x16x32_f16(false, af.v, false, b1,
                                                  (short)0, acc1, false, false);
    acc2 = __builtin_amdgcn_wmma_f32_16x16x32_f16(false, af.v, false, b2,
                                                  (short)0, acc2, false, false);
    acc3 = __builtin_amdgcn_wmma_f32_16x16x32_f16(false, af.v, false, b3,
                                                  (short)0, acc3, false, false);
  }

  // Epilogue (guards only here): VGPR r -> row m0 + half_id*8 + r, col n0+16t+lr
  v8f accs[4] = {acc0, acc1, acc2, acc3};
#pragma unroll
  for (int t = 0; t < 4; ++t) {
    const int col = n0 + t * 16 + lr;
    if (col >= N) continue;
    const float bv = bias ? bias[col] : 0.0f;
#pragma unroll
    for (int r = 0; r < 8; ++r) {
      const int row = m0 + half_id * 8 + r;
      if (row < M) {
        float v = accs[t][r] + bv;
        if (addend) v += addend[(size_t)row * N + col];
        if (relu_flag) v = fmaxf(v, 0.0f);
        C[(size_t)row * N + col] = v;
      }
    }
  }
}

// ---------------------------------------------------------------------------
// fp32 [M,K] -> zero-padded f16 [Mp,Kp]
// ---------------------------------------------------------------------------
__global__ void cvt_pad_f16_kernel(const float* __restrict__ src,
                                   _Float16* __restrict__ dst,
                                   int M, int K, int Mp, int Kp) {
  const int i = blockIdx.x * blockDim.x + threadIdx.x;
  if (i >= Mp * Kp) return;
  const int r = i / Kp;
  const int c = i - r * Kp;
  dst[i] = (r < M && c < K) ? (_Float16)src[(size_t)r * K + c] : (_Float16)0.0f;
}

__global__ void fill_zero_f16_kernel(_Float16* __restrict__ p, int n) {
  const int i = blockIdx.x * blockDim.x + threadIdx.x;
  if (i < n) p[i] = (_Float16)0.0f;
}

// ---------------------------------------------------------------------------
// message_atom += sum_j(nei) * max_j(nei)   (fp32, in place)
// ---------------------------------------------------------------------------
__global__ void agg_add_kernel(const float* __restrict__ mbond,
                               const int* __restrict__ a2b,
                               float* __restrict__ matom) {
  const int i = blockIdx.x * blockDim.x + threadIdx.x;
  if (i >= N_ATOMS * H_DIM) return;
  const int a = i / H_DIM;
  const int h = i - a * H_DIM;
  float s = 0.0f, mx = -3.0e38f;
#pragma unroll
  for (int j = 0; j < MAX_NB; ++j) {
    const int e = a2b[a * MAX_NB + j];
    const float v = mbond[(size_t)e * H_DIM + h];
    s += v;
    mx = fmaxf(mx, v);
  }
  matom[i] += s * mx;
}

// tmpE16[e,:] = f16( message_atom[b2a[e]] - message_bond[b2revb[e]] ), padded
__global__ void bond_diff_f16_kernel(const float* __restrict__ matom,
                                     const float* __restrict__ mbond,
                                     const int* __restrict__ b2a,
                                     const int* __restrict__ b2revb,
                                     _Float16* __restrict__ dst) {
  const int i = blockIdx.x * blockDim.x + threadIdx.x;
  if (i >= E_PAD * K300P) return;
  const int e = i / K300P;
  const int c = i - e * K300P;
  float v = 0.0f;
  if (e < N_BONDS && c < H_DIM) {
    const int a  = b2a[e];
    const int re = b2revb[e];
    v = matom[(size_t)a * H_DIM + c] - mbond[(size_t)re * H_DIM + c];
  }
  dst[i] = (_Float16)v;
}

// concat16[a, 0:300]=agg  [300:600]=message_atom  [600:900]=input_atom, padded
__global__ void build_concat16_kernel(const float* __restrict__ mbond,
                                      const int* __restrict__ a2b,
                                      const float* __restrict__ matom,
                                      const float* __restrict__ iatom,
                                      _Float16* __restrict__ cc) {
  const int i = blockIdx.x * blockDim.x + threadIdx.x;
  if (i >= A_PAD * K900P) return;
  const int a = i / K900P;
  const int c = i - a * K900P;
  float v = 0.0f;
  if (a < N_ATOMS) {
    if (c < H_DIM) {
      float s = 0.0f, mx = -3.0e38f;
#pragma unroll
      for (int j = 0; j < MAX_NB; ++j) {
        const int e = a2b[a * MAX_NB + j];
        const float x = mbond[(size_t)e * H_DIM + c];
        s += x;
        mx = fmaxf(mx, x);
      }
      v = s * mx;
    } else if (c < 2 * H_DIM) {
      v = matom[(size_t)a * H_DIM + (c - H_DIM)];
    } else if (c < 3 * H_DIM) {
      v = iatom[(size_t)a * H_DIM + (c - 2 * H_DIM)];
    }
  }
  cc[i] = (_Float16)v;
}

// message = relu(node + gru_bias)
__global__ void bias_relu_kernel(const float* __restrict__ node,
                                 const float* __restrict__ bias,
                                 float* __restrict__ out, int total) {
  const int i = blockIdx.x * blockDim.x + threadIdx.x;
  if (i >= total) return;
  out[i] = fmaxf(node[i] + bias[i % H_DIM], 0.0f);
}

// padded16[b,l,:] (f16, K-padded); h0/hf/hb fp32; hf16/hb16 f16 K-padded
__global__ void pad_h0_kernel(const float* __restrict__ message,
                              const float* __restrict__ node,
                              const int* __restrict__ a_start,
                              const int* __restrict__ a_size,
                              _Float16* __restrict__ padded16,
                              float* __restrict__ h0,
                              float* __restrict__ hf, float* __restrict__ hb,
                              _Float16* __restrict__ hf16,
                              _Float16* __restrict__ hb16) {
  const int i = blockIdx.x * blockDim.x + threadIdx.x;
  if (i >= N_MOLS * K300P) return;
  const int b = i / K300P;
  const int c = i - b * K300P;
  const bool act = c < H_DIM;
  const int st = a_start[b];
  const int sz = a_size[b];
  float mx = -1.0e30f;
  for (int l = 0; l < L_SEQ; ++l) {
    const bool mk = act && (l < sz);
    const int a = mk ? (st + l) : 0;
    const float pv = mk ? message[(size_t)a * H_DIM + c] : 0.0f;
    padded16[((size_t)(b * L_SEQ + l)) * K300P + c] = (_Float16)pv;
    const float nv = mk ? node[(size_t)a * H_DIM + c] : -1.0e30f;
    mx = fmaxf(mx, nv);
  }
  if (act) {
    h0[b * H_DIM + c] = mx;
    hf[b * H_DIM + c] = mx;
    hb[b * H_DIM + c] = mx;
  }
  hf16[i] = act ? (_Float16)mx : (_Float16)0.0f;
  hb16[i] = act ? (_Float16)mx : (_Float16)0.0f;
}

__device__ __forceinline__ float sigmoidf_(float x) {
  return 1.0f / (1.0f + __expf(-x));
}

// One GRU step (gate order r,z,n); also refreshes f16-staged h for next GEMM.
__global__ void gru_gate_kernel(const float* __restrict__ gi,   // [B,L,3H]
                                const float* __restrict__ gh,   // [B,3H]
                                float* __restrict__ h,          // [B,H] in/out
                                _Float16* __restrict__ h16,     // [B,K300P]
                                float* __restrict__ out,        // [B,L,H]
                                int tcol) {
  const int i = blockIdx.x * blockDim.x + threadIdx.x;
  if (i >= N_MOLS * H_DIM) return;
  const int b  = i / H_DIM;
  const int hh = i - b * H_DIM;
  const size_t gir = ((size_t)(b * L_SEQ + tcol)) * H3_DIM;
  const size_t ghr = (size_t)b * H3_DIM;
  const float r = sigmoidf_(gi[gir + hh] + gh[ghr + hh]);
  const float z = sigmoidf_(gi[gir + H_DIM + hh] + gh[ghr + H_DIM + hh]);
  const float n = tanhf(gi[gir + 2 * H_DIM + hh] + r * gh[ghr + 2 * H_DIM + hh]);
  const float hnew = (1.0f - z) * n + z * h[i];
  h[i] = hnew;
  h16[(size_t)b * K300P + hh] = (_Float16)hnew;
  out[((size_t)(b * L_SEQ + tcol)) * H_DIM + hh] = hnew;
}

// full16[idx[b,l], 0:H]=out_f ; [H:2H]=out_b   (f16, row/col padded)
__global__ void scatter_full16_kernel(const float* __restrict__ out_f,
                                      const float* __restrict__ out_b,
                                      const int* __restrict__ a_start,
                                      const int* __restrict__ a_size,
                                      _Float16* __restrict__ full16) {
  const int i = blockIdx.x * blockDim.x + threadIdx.x;
  if (i >= N_MOLS * L_SEQ * H_DIM) return;
  const int h  = i % H_DIM;
  const int bl = i / H_DIM;
  const int l  = bl % L_SEQ;
  const int b  = bl / L_SEQ;
  if (l >= a_size[b]) return;
  const int a = a_start[b] + l;
  full16[(size_t)a * K600P + h]         = (_Float16)out_f[i];
  full16[(size_t)a * K600P + H_DIM + h] = (_Float16)out_b[i];
}

__global__ void full_row0_kernel(const float* __restrict__ message,
                                 _Float16* __restrict__ full16) {
  const int h = blockIdx.x * blockDim.x + threadIdx.x;
  if (h < H_DIM) {
    full16[h]         = (_Float16)message[h];
    full16[H_DIM + h] = (_Float16)message[h];
  }
}

// mol_vecs[b,:] = mean over valid atoms of atom_h[idx]
__global__ void pool_kernel(const float* __restrict__ atom_h,
                            const int* __restrict__ a_start,
                            const int* __restrict__ a_size,
                            float* __restrict__ out) {
  const int i = blockIdx.x * blockDim.x + threadIdx.x;
  if (i >= N_MOLS * H_DIM) return;
  const int b = i / H_DIM;
  const int h = i - b * H_DIM;
  const int st = a_start[b];
  const int sz = a_size[b];
  float s = 0.0f;
  for (int l = 0; l < L_SEQ; ++l)
    if (l < sz) s += atom_h[(size_t)(st + l) * H_DIM + h];
  out[i] = s / (float)sz;
}

// ---------------------------------------------------------------------------
// Host side
// ---------------------------------------------------------------------------
static inline size_t align256(size_t x) { return (x + 255) & ~(size_t)255; }

static inline void launch_gemm(const _Float16* A16, int lda,
                               const _Float16* W16, int ldw,
                               const float* bias, const float* addend,
                               float* C, int M, int N, int Kp, int relu,
                               hipStream_t s) {
  dim3 grid((N + 63) / 64, (M + 63) / 64);
  wmma_gemm2_kernel<<<grid, 128, 0, s>>>(A16, lda, W16, ldw, bias, addend, C,
                                         M, N, Kp, relu);
}

static inline void launch_cvt(const float* src, _Float16* dst, int M, int K,
                              int Mp, int Kp, hipStream_t s) {
  const int n = Mp * Kp;
  cvt_pad_f16_kernel<<<(n + 255) / 256, 256, 0, s>>>(src, dst, M, K, Mp, Kp);
}

extern "C" void kernel_launch(void* const* d_in, const int* in_sizes, int n_in,
                              void* d_out, int out_size, void* d_ws, size_t ws_size,
                              hipStream_t stream) {
  const float* f_atoms  = (const float*)d_in[0];
  const float* f_bonds  = (const float*)d_in[1];
  const int*   a2b      = (const int*)d_in[2];
  const int*   b2a      = (const int*)d_in[3];
  const int*   b2revb   = (const int*)d_in[4];
  const int*   a_start  = (const int*)d_in[5];
  const int*   a_size   = (const int*)d_in[6];
  const float* W_i_atom = (const float*)d_in[7];
  const float* W_i_bond = (const float*)d_in[8];
  const float* W_h      = (const float*)d_in[9];
  const float* lr_W     = (const float*)d_in[10];
  const float* W_o_W    = (const float*)d_in[11];
  const float* gru_bias = (const float*)d_in[12];
  const float* Wih_f    = (const float*)d_in[13];
  const float* Whh_f    = (const float*)d_in[14];
  const float* bih_f    = (const float*)d_in[15];
  const float* bhh_f    = (const float*)d_in[16];
  const float* Wih_b    = (const float*)d_in[17];
  const float* Whh_b    = (const float*)d_in[18];
  const float* bih_b    = (const float*)d_in[19];
  const float* bhh_b    = (const float*)d_in[20];

  const int A  = N_ATOMS;
  const int E  = N_BONDS;
  const int Hd = H_DIM;
  const int B  = N_MOLS;
  const int Ls = L_SEQ;
  const int BL = BL_SEQ;

  // ---- workspace bump allocator --------------------------------------------
  char* ws = (char*)d_ws;
  size_t off = 0;
  auto allocb = [&](size_t bytes) -> char* {
    char* p = ws + off;
    off = align256(off + bytes);
    return p;
  };
  auto allocf   = [&](size_t n) -> float*    { return (float*)allocb(n * sizeof(float)); };
  auto alloch16 = [&](size_t n) -> _Float16* { return (_Float16*)allocb(n * sizeof(_Float16)); };

  // f16 weights, [Npad64 x Kpad32] zero-padded
  _Float16* w_i_atom16 = alloch16((size_t)N300P * K133P);
  _Float16* w_i_bond16 = alloch16((size_t)N300P * K147P);
  _Float16* w_h16      = alloch16((size_t)2 * N300P * K300P);
  _Float16* lr_w16     = alloch16((size_t)N300P * K900P);
  _Float16* w_o16      = alloch16((size_t)N300P * K600P);
  _Float16* wih_f16    = alloch16((size_t)N900P * K300P);
  _Float16* whh_f16    = alloch16((size_t)N900P * K300P);
  _Float16* wih_b16    = alloch16((size_t)N900P * K300P);
  _Float16* whh_b16    = alloch16((size_t)N900P * K300P);

  // f16 staged GEMM A-operands
  _Float16* f_atoms16 = alloch16((size_t)A_PAD * K133P);
  _Float16* f_bonds16 = alloch16((size_t)E_PAD * K147P);
  _Float16* tmpE16    = alloch16((size_t)E_PAD * K300P);
  _Float16* concat16  = alloch16((size_t)A_PAD * K900P);
  _Float16* padded16  = alloch16((size_t)BL * K300P);
  _Float16* hf16      = alloch16((size_t)B * K300P);
  _Float16* hb16      = alloch16((size_t)B * K300P);
  _Float16* full16    = alloch16((size_t)A_PAD * K600P);

  // fp32 activations
  float* input_atom   = allocf((size_t)A * Hd);
  float* message_atom = allocf((size_t)A * Hd);
  float* input_bond   = allocf((size_t)E * Hd);
  float* message_bond = allocf((size_t)E * Hd);
  float* node         = allocf((size_t)A * Hd);
  float* message      = allocf((size_t)A * Hd);
  float* h0buf        = allocf((size_t)B * Hd);
  float* hf           = allocf((size_t)B * Hd);
  float* hb           = allocf((size_t)B * Hd);
  float* gi_f         = allocf((size_t)BL * H3_DIM);
  float* gi_b         = allocf((size_t)BL * H3_DIM);
  float* gh_f         = allocf((size_t)B * H3_DIM);
  float* gh_b         = allocf((size_t)B * H3_DIM);
  float* out_f        = allocf((size_t)BL * Hd);
  float* out_b        = allocf((size_t)BL * Hd);
  float* atom_h       = allocf((size_t)A * Hd);
  (void)ws_size;

  // ---- stage weights + inputs to padded f16 --------------------------------
  launch_cvt(W_i_atom, w_i_atom16, Hd, 133, N300P, K133P, stream);
  launch_cvt(W_i_bond, w_i_bond16, Hd, 147, N300P, K147P, stream);
  launch_cvt(W_h,               w_h16,               Hd, Hd, N300P, K300P, stream);
  launch_cvt(W_h + 90000,       w_h16 + N300P*K300P, Hd, Hd, N300P, K300P, stream);
  launch_cvt(lr_W,  lr_w16, Hd, H3_DIM, N300P, K900P, stream);
  launch_cvt(W_o_W, w_o16,  Hd, 2 * Hd, N300P, K600P, stream);
  launch_cvt(Wih_f, wih_f16, H3_DIM, Hd, N900P, K300P, stream);
  launch_cvt(Whh_f, whh_f16, H3_DIM, Hd, N900P, K300P, stream);
  launch_cvt(Wih_b, wih_b16, H3_DIM, Hd, N900P, K300P, stream);
  launch_cvt(Whh_b, whh_b16, H3_DIM, Hd, N900P, K300P, stream);
  launch_cvt(f_atoms, f_atoms16, A, 133, A_PAD, K133P, stream);
  launch_cvt(f_bonds, f_bonds16, E, 147, E_PAD, K147P, stream);

  // ---- input projections ---------------------------------------------------
  launch_gemm(f_atoms16, K133P, w_i_atom16, K133P, nullptr, nullptr,
              input_atom, A, Hd, K133P, 1, stream);
  launch_gemm(f_bonds16, K147P, w_i_bond16, K147P, nullptr, nullptr,
              input_bond, E, Hd, K147P, 1, stream);
  hipMemcpyAsync(message_atom, input_atom, (size_t)A * Hd * sizeof(float),
                 hipMemcpyDeviceToDevice, stream);
  hipMemcpyAsync(message_bond, input_bond, (size_t)E * Hd * sizeof(float),
                 hipMemcpyDeviceToDevice, stream);

  // ---- depth loop ----------------------------------------------------------
  const int aggBlocks   = (A * Hd + 255) / 256;
  const int diffBlocks  = (E_PAD * K300P + 255) / 256;
  for (int d = 0; d < 2; ++d) {
    agg_add_kernel<<<aggBlocks, 256, 0, stream>>>(message_bond, a2b, message_atom);
    bond_diff_f16_kernel<<<diffBlocks, 256, 0, stream>>>(
        message_atom, message_bond, b2a, b2revb, tmpE16);
    launch_gemm(tmpE16, K300P, w_h16 + (size_t)d * N300P * K300P, K300P,
                nullptr, input_bond, message_bond, E, Hd, K300P, 1, stream);
  }

  // ---- final aggregation + node projection --------------------------------
  const int ccBlocks = (A_PAD * K900P + 255) / 256;
  build_concat16_kernel<<<ccBlocks, 256, 0, stream>>>(
      message_bond, a2b, message_atom, input_atom, concat16);
  launch_gemm(concat16, K900P, lr_w16, K900P, nullptr, nullptr,
              node, A, Hd, K900P, 0, stream);
  bias_relu_kernel<<<aggBlocks, 256, 0, stream>>>(node, gru_bias, message, A * Hd);

  // ---- BiGRU ---------------------------------------------------------------
  const int padBlocks = (B * K300P + 255) / 256;
  const int bhBlocks  = (B * Hd + 255) / 256;
  pad_h0_kernel<<<padBlocks, 256, 0, stream>>>(message, node, a_start, a_size,
                                               padded16, h0buf, hf, hb, hf16, hb16);
  launch_gemm(padded16, K300P, wih_f16, K300P, bih_f, nullptr,
              gi_f, BL, H3_DIM, K300P, 0, stream);
  launch_gemm(padded16, K300P, wih_b16, K300P, bih_b, nullptr,
              gi_b, BL, H3_DIM, K300P, 0, stream);

  for (int t = 0; t < Ls; ++t) {
    launch_gemm(hf16, K300P, whh_f16, K300P, bhh_f, nullptr,
                gh_f, B, H3_DIM, K300P, 0, stream);
    gru_gate_kernel<<<bhBlocks, 256, 0, stream>>>(gi_f, gh_f, hf, hf16, out_f, t);
    launch_gemm(hb16, K300P, whh_b16, K300P, bhh_b, nullptr,
                gh_b, B, H3_DIM, K300P, 0, stream);
    gru_gate_kernel<<<bhBlocks, 256, 0, stream>>>(gi_b, gh_b, hb, hb16, out_b,
                                                  Ls - 1 - t);
  }

  // ---- scatter back + output head -----------------------------------------
  const int fullN = A_PAD * K600P;
  fill_zero_f16_kernel<<<(fullN + 255) / 256, 256, 0, stream>>>(full16, fullN);
  scatter_full16_kernel<<<(BL * Hd + 255) / 256, 256, 0, stream>>>(
      out_f, out_b, a_start, a_size, full16);
  full_row0_kernel<<<(Hd + 255) / 256, 256, 0, stream>>>(message, full16);
  launch_gemm(full16, K600P, w_o16, K600P, nullptr, nullptr,
              atom_h, A, Hd, K600P, 1, stream);

  pool_kernel<<<bhBlocks, 256, 0, stream>>>(atom_h, a_start, a_size,
                                            (float*)d_out);
  (void)in_sizes; (void)n_in; (void)out_size;
}
